// GATLayer_32152125178166
// MI455X (gfx1250) — matrix operand order repaired and checked by reading, not compiled, not run on hardware
//
#include <hip/hip_runtime.h>

// ---------------------------------------------------------------------------
// Problem constants (from reference)
// ---------------------------------------------------------------------------
constexpr int NN    = 10000;           // nodes
constexpr int EE    = 160000;          // raw edges
constexpr int ET    = EE + NN;         // edges + self loops
constexpr int HEADS = 4;
constexpr int HID   = 256;
constexpr int F1    = HEADS * HID;     // 1024
constexpr int IN_DIM  = 512;
constexpr int OUT_DIM = 512;
constexpr float NEG_SLOPE = 0.2f;

// ---------------------------------------------------------------------------
// WMMA types (CDNA5 / gfx1250, wave32)
// ---------------------------------------------------------------------------
typedef __attribute__((ext_vector_type(16))) __bf16 v16bf;
typedef __attribute__((ext_vector_type(8)))  float  v8f;

union FragBF { uint4 u[2]; v16bf v; };   // 32 bytes = 8 VGPRs

__device__ __forceinline__ unsigned short f2bf(float f) {
  unsigned int u = __float_as_uint(f);
  u += 0x7FFFu + ((u >> 16) & 1u);       // round-to-nearest-even
  return (unsigned short)(u >> 16);
}

// ---------------------------------------------------------------------------
// GEMM: C[M,N] = A[M,K] @ B[K,N] (+bias), fp32 in/out, bf16 WMMA compute.
// Block tile 128x128, BK=32, 256 threads = 8 waves (2 x 4), wave tile 64x32.
// ---------------------------------------------------------------------------
#define BM 128
#define BN 128
#define BK 32
#define LDT 40   // LDS row stride in ushorts (80B: 16B-aligned, conflict-skewed)

__global__ __launch_bounds__(256) void gemm_wmma_bf16(
    const float* __restrict__ A, const float* __restrict__ B,
    const float* __restrict__ bias, float* __restrict__ C,
    int M, int K, int N) {
  __shared__ __align__(16) unsigned short As[BM * LDT];
  __shared__ __align__(16) unsigned short Bs[BN * LDT];

  const int bm = blockIdx.x * BM;
  const int bn = blockIdx.y * BN;
  const int t    = threadIdx.x;
  const int wave = t >> 5;
  const int lane = t & 31;
  const int wm   = wave >> 2;        // 0..1  (M direction, 64 rows)
  const int wn   = wave & 3;         // 0..3  (N direction, 32 cols)
  const int half = lane >> 4;        // K-half select per ISA 16-bit layouts
  const int l16  = lane & 15;

  v8f acc[4][2] = {};

  for (int k0 = 0; k0 < K; k0 += BK) {
    // ---- stage A tile (128x32 fp32 -> bf16), row-major ----
#pragma unroll
    for (int i = 0; i < 4; ++i) {
      int idx = t + 256 * i;               // 1024 float4 total
      int row = idx >> 3, c4 = idx & 7;
      int gm  = bm + row;
      float4 v = make_float4(0.f, 0.f, 0.f, 0.f);
      if (gm < M) {
        const float* gp = A + (size_t)gm * K + k0 + c4 * 4;
        v = *(const float4*)gp;
        if (k0 + BK < K) __builtin_prefetch(gp + BK, 0, 1);   // global_prefetch_b8
      }
      ushort4 u;
      u.x = f2bf(v.x); u.y = f2bf(v.y); u.z = f2bf(v.z); u.w = f2bf(v.w);
      *(ushort4*)(&As[row * LDT + c4 * 4]) = u;
    }
    // ---- stage B tile (32x128 fp32) transposed -> Bs[n][k] bf16 ----
#pragma unroll
    for (int i = 0; i < 4; ++i) {
      int idx = t + 256 * i;
      int kr = idx >> 5, c4 = idx & 31;
      const float* gp = B + (size_t)(k0 + kr) * N + bn + c4 * 4;
      float4 v = *(const float4*)gp;
      if (k0 + BK < K) __builtin_prefetch(gp + (size_t)BK * N, 0, 1);
      Bs[(c4 * 4 + 0) * LDT + kr] = f2bf(v.x);
      Bs[(c4 * 4 + 1) * LDT + kr] = f2bf(v.y);
      Bs[(c4 * 4 + 2) * LDT + kr] = f2bf(v.z);
      Bs[(c4 * 4 + 3) * LDT + kr] = f2bf(v.w);
    }
    __syncthreads();

    // ---- B fragments: lane holds N=l16, K = 8*half..8*half+7, 16+8*half..16+8*half+7 ----
    FragBF bf[2];
#pragma unroll
    for (int fn = 0; fn < 2; ++fn) {
      int n = wn * 32 + fn * 16 + l16;
      const uint4* p = (const uint4*)(&Bs[n * LDT]);
      bf[fn].u[0] = p[half];
      bf[fn].u[1] = p[2 + half];
    }
    // ---- A fragments + WMMA ----
#pragma unroll
    for (int fm = 0; fm < 4; ++fm) {
      int m = wm * 64 + fm * 16 + l16;
      const uint4* p = (const uint4*)(&As[m * LDT]);
      FragBF af;
      af.u[0] = p[half];         // K = 8*half .. 8*half+7
      af.u[1] = p[2 + half];     // K = 16+8*half .. 16+8*half+7
#pragma unroll
      for (int fn = 0; fn < 2; ++fn) {
        acc[fm][fn] = __builtin_amdgcn_wmma_f32_16x16x32_bf16(
            false, af.v, false, bf[fn].v, (short)0, acc[fm][fn], false, false);
      }
    }
    __syncthreads();
  }

  // ---- epilogue: C/D layout -> lane N=l16, VGPR r -> M = r + 8*half ----
#pragma unroll
  for (int fm = 0; fm < 4; ++fm) {
#pragma unroll
    for (int fn = 0; fn < 2; ++fn) {
      int gn = bn + wn * 32 + fn * 16 + l16;
      float bv = bias ? bias[gn] : 0.f;
#pragma unroll
      for (int r = 0; r < 8; ++r) {
        int gm = bm + wm * 64 + fm * 16 + r + 8 * half;
        if (gm < M) C[(size_t)gm * N + gn] = acc[fm][fn][r] + bv;
      }
    }
  }
}

// ---------------------------------------------------------------------------
// Attention coefficients: a_src[n,h] = sum_c h[n,h,c]*att_src[h,c]  (one wave
// per (node, head); coalesced lane-strided loads + butterfly reduce)
// ---------------------------------------------------------------------------
__global__ void attn_coef_kernel(const float* __restrict__ h,
                                 const float* __restrict__ att_s,
                                 const float* __restrict__ att_d,
                                 float* __restrict__ a_s,
                                 float* __restrict__ a_d) {
  int gw   = (int)((blockIdx.x * blockDim.x + threadIdx.x) >> 5);
  int lane = threadIdx.x & 31;
  if (gw >= NN * HEADS) return;
  int node = gw >> 2, hd = gw & 3;
  const float* hp  = h + (size_t)node * F1 + hd * HID;
  const float* asp = att_s + hd * HID;
  const float* adp = att_d + hd * HID;
  float vs = 0.f, vd = 0.f;
#pragma unroll
  for (int c = lane; c < HID; c += 32) {
    float hv = hp[c];
    vs += hv * asp[c];
    vd += hv * adp[c];
  }
#pragma unroll
  for (int off = 16; off; off >>= 1) {
    vs += __shfl_xor(vs, off, 32);
    vd += __shfl_xor(vd, off, 32);
  }
  if (lane == 0) { a_s[gw] = vs; a_d[gw] = vd; }
}

// ---------------------------------------------------------------------------
// Edge helpers (self loops appended implicitly: e >= EE -> (i,i))
// ---------------------------------------------------------------------------
__device__ __forceinline__ void edge_sd(const int* __restrict__ ei, int e,
                                        int& s, int& d) {
  if (e < EE) { s = ei[e]; d = ei[EE + e]; }
  else        { s = d = e - EE; }
}

__device__ __forceinline__ void atomicMaxF(float* addr, float v) {
  if (v >= 0.f) atomicMax((int*)addr, __float_as_int(v));
  else          atomicMin((unsigned int*)addr, __float_as_uint(v));
}

__global__ void fill_kernel(float* __restrict__ p, long n, float v) {
  long i = (long)blockIdx.x * blockDim.x + threadIdx.x;
  long stride = (long)gridDim.x * blockDim.x;
  for (; i < n; i += stride) p[i] = v;
}

// logit = leaky_relu(a_src[src] + a_dst[dst]); segment max over dst
__global__ void edge_max_kernel(const int* __restrict__ ei,
                                const float* __restrict__ a_s,
                                const float* __restrict__ a_d,
                                float* __restrict__ logit,
                                float* __restrict__ mx) {
  int tid = blockIdx.x * blockDim.x + threadIdx.x;
  if (tid >= ET * HEADS) return;
  int e = tid >> 2, hd = tid & 3;
  int s, d; edge_sd(ei, e, s, d);
  float v = a_s[s * HEADS + hd] + a_d[d * HEADS + hd];
  v = v > 0.f ? v : NEG_SLOPE * v;
  logit[tid] = v;
  atomicMaxF(&mx[d * HEADS + hd], v);
}

__global__ void edge_expsum_kernel(const int* __restrict__ ei,
                                   float* __restrict__ alpha,
                                   const float* __restrict__ mx,
                                   float* __restrict__ ssum) {
  int tid = blockIdx.x * blockDim.x + threadIdx.x;
  if (tid >= ET * HEADS) return;
  int e = tid >> 2, hd = tid & 3;
  int s, d; edge_sd(ei, e, s, d);
  (void)s;
  float w = __expf(alpha[tid] - mx[d * HEADS + hd]);
  alpha[tid] = w;
  atomicAdd(&ssum[d * HEADS + hd], w);
}

__global__ void edge_norm_kernel(const int* __restrict__ ei,
                                 float* __restrict__ alpha,
                                 const float* __restrict__ ssum) {
  int tid = blockIdx.x * blockDim.x + threadIdx.x;
  if (tid >= ET * HEADS) return;
  int e = tid >> 2, hd = tid & 3;
  int s, d; edge_sd(ei, e, s, d);
  (void)s;
  alpha[tid] = alpha[tid] / (ssum[d * HEADS + hd] + 1e-16f);
}

// out[dst] += alpha * h[src]; one edge per 256-thread block, float4 gathers
__global__ void scatter_kernel(const int* __restrict__ ei,
                               const float* __restrict__ alpha,
                               const float* __restrict__ h,
                               float* __restrict__ out) {
  int e = blockIdx.x;
  int t = threadIdx.x;                 // channels [4t, 4t+3] of 1024
  int s, d; edge_sd(ei, e, s, d);
  int hd = t >> 6;                     // 4*t / 256
  float al = alpha[e * HEADS + hd];
  float4 hv = *(const float4*)(h + (size_t)s * F1 + t * 4);
  float* op = out + (size_t)d * F1 + t * 4;
  atomicAdd(op + 0, al * hv.x);
  atomicAdd(op + 1, al * hv.y);
  atomicAdd(op + 2, al * hv.z);
  atomicAdd(op + 3, al * hv.w);
}

__global__ void bias_relu_kernel(float* __restrict__ p,
                                 const float* __restrict__ bias, long n) {
  long i = (long)blockIdx.x * blockDim.x + threadIdx.x;
  long stride = (long)gridDim.x * blockDim.x;
  for (; i < n; i += stride) {
    float v = p[i] + bias[i & (F1 - 1)];
    p[i] = v > 0.f ? v : 0.f;
  }
}

// ---------------------------------------------------------------------------
// Driver
// ---------------------------------------------------------------------------
extern "C" void kernel_launch(void* const* d_in, const int* in_sizes, int n_in,
                              void* d_out, int out_size, void* d_ws, size_t ws_size,
                              hipStream_t stream) {
  const float* x   = (const float*)d_in[0];
  const int*   ei  = (const int*)d_in[1];
  const float* W1  = (const float*)d_in[2];
  const float* as1 = (const float*)d_in[3];
  const float* ad1 = (const float*)d_in[4];
  const float* b1  = (const float*)d_in[5];
  const float* W2  = (const float*)d_in[6];
  const float* as2 = (const float*)d_in[7];
  const float* ad2 = (const float*)d_in[8];
  const float* b2  = (const float*)d_in[9];
  const float* fcW = (const float*)d_in[10];
  const float* fcb = (const float*)d_in[11];
  float* out = (float*)d_out;

  // workspace layout (floats)
  float* H  = (float*)d_ws;            // NN*F1  feature buffer (h1, then h2)
  float* G  = H  + (size_t)NN * F1;    // NN*F1  aggregation buffer
  float* AS = G  + (size_t)NN * F1;    // NN*HEADS
  float* AD = AS + NN * HEADS;
  float* MX = AD + NN * HEADS;
  float* SS = MX + NN * HEADS;
  float* AL = SS + NN * HEADS;         // ET*HEADS edge logits/weights/alpha

  const dim3 blk(256);
  const int gridM = (NN + BM - 1) / BM;          // 79
  const int egrid = (ET * HEADS + 255) / 256;
  const long nF   = (long)NN * F1;

  // ---------------- layer 1 ----------------
  gemm_wmma_bf16<<<dim3(gridM, F1 / BN), blk, 0, stream>>>(x, W1, nullptr, H,
                                                           NN, IN_DIM, F1);
  attn_coef_kernel<<<(NN * HEADS * 32 + 255) / 256, blk, 0, stream>>>(H, as1, ad1, AS, AD);
  fill_kernel<<<64,   blk, 0, stream>>>(MX, NN * HEADS, -1e30f);
  fill_kernel<<<64,   blk, 0, stream>>>(SS, NN * HEADS, 0.f);
  fill_kernel<<<2048, blk, 0, stream>>>(G, nF, 0.f);
  edge_max_kernel<<<egrid, blk, 0, stream>>>(ei, AS, AD, AL, MX);
  edge_expsum_kernel<<<egrid, blk, 0, stream>>>(ei, AL, MX, SS);
  edge_norm_kernel<<<egrid, blk, 0, stream>>>(ei, AL, SS);
  scatter_kernel<<<ET, blk, 0, stream>>>(ei, AL, H, G);
  bias_relu_kernel<<<2048, blk, 0, stream>>>(G, b1, nF);

  // ---------------- layer 2 ----------------
  gemm_wmma_bf16<<<dim3(gridM, F1 / BN), blk, 0, stream>>>(G, W2, nullptr, H,
                                                           NN, F1, F1);
  attn_coef_kernel<<<(NN * HEADS * 32 + 255) / 256, blk, 0, stream>>>(H, as2, ad2, AS, AD);
  fill_kernel<<<64,   blk, 0, stream>>>(MX, NN * HEADS, -1e30f);
  fill_kernel<<<64,   blk, 0, stream>>>(SS, NN * HEADS, 0.f);
  fill_kernel<<<2048, blk, 0, stream>>>(G, nF, 0.f);     // GEMM2 done reading G
  edge_max_kernel<<<egrid, blk, 0, stream>>>(ei, AS, AD, AL, MX);
  edge_expsum_kernel<<<egrid, blk, 0, stream>>>(ei, AL, MX, SS);
  edge_norm_kernel<<<egrid, blk, 0, stream>>>(ei, AL, SS);
  scatter_kernel<<<ET, blk, 0, stream>>>(ei, AL, H, G);
  bias_relu_kernel<<<2048, blk, 0, stream>>>(G, b2, nF);

  // ---------------- final FC ----------------
  gemm_wmma_bf16<<<dim3(gridM, OUT_DIM / BN), blk, 0, stream>>>(G, fcW, fcb, out,
                                                                NN, F1, OUT_DIM);
}
